// StreamLM_82094004896063
// MI455X (gfx1250) — matrix operand order repaired and checked by reading, not compiled, not run on hardware
//
#include <hip/hip_runtime.h>

// ---------------------------------------------------------------------------
// Problem constants (match reference)
// ---------------------------------------------------------------------------
#define VOCAB 32000
#define DMODEL 1024
#define BATCH 2
#define TSEQ 1024
#define MTOT (BATCH * TSEQ)     // 2048
#define LEAK 0.95f

// ---------------------------------------------------------------------------
// CDNA5 WMMA types
// ---------------------------------------------------------------------------
typedef __attribute__((ext_vector_type(16))) __bf16 v16bf;
typedef __attribute__((ext_vector_type(8)))  __bf16 v8bf;
typedef __attribute__((ext_vector_type(8)))  float  v8f;
typedef __attribute__((ext_vector_type(4)))  unsigned short us4;

// fp32 -> bf16 round-to-nearest-even
static __device__ __forceinline__ unsigned short f2bf(float f) {
    unsigned u = __float_as_uint(f);
    u += 0x7FFFu + ((u >> 16) & 1u);
    return (unsigned short)(u >> 16);
}

// A fragment: lane holds row m, contiguous K = {k+16h .. +15}
static __device__ __forceinline__ v16bf load_fragA(const unsigned short* p) {
    v8bf lo = *reinterpret_cast<const v8bf*>(p);
    v8bf hi = *reinterpret_cast<const v8bf*>(p + 8);
    return __builtin_shufflevector(lo, hi, 0,1,2,3,4,5,6,7,8,9,10,11,12,13,14,15);
}
// B fragment: lane holds weight row n, contiguous K = {k+16h .. +15}
static __device__ __forceinline__ v16bf load_fragB(const unsigned short* p) {
    return *reinterpret_cast<const v16bf*>(p);
}

// ---------------------------------------------------------------------------
// 1) fp32 -> bf16 conversion (n divisible by 4)
// ---------------------------------------------------------------------------
__global__ __launch_bounds__(256) void cvt_bf16_kernel(const float* __restrict__ src,
                                                       unsigned short* __restrict__ dst,
                                                       int n) {
    int stride = gridDim.x * blockDim.x * 4;
    for (int i = (blockIdx.x * blockDim.x + threadIdx.x) * 4; i < n; i += stride) {
        float4 f = *reinterpret_cast<const float4*>(src + i);
        us4 o = { f2bf(f.x), f2bf(f.y), f2bf(f.z), f2bf(f.w) };
        *reinterpret_cast<us4*>(dst + i) = o;
    }
}

// ---------------------------------------------------------------------------
// 2) gather + positional embedding + LayerNorm -> bf16 x-hat  (block per row)
// ---------------------------------------------------------------------------
__global__ __launch_bounds__(256) void embed_ln_kernel(const int* __restrict__ idx,
                                                       const float* __restrict__ tokw,
                                                       const float* __restrict__ posw,
                                                       const float* __restrict__ g,
                                                       const float* __restrict__ bln,
                                                       unsigned short* __restrict__ xb) {
    __shared__ float s_sum[256];
    __shared__ float s_sq[256];
    const int row = blockIdx.x;             // 0..MTOT-1
    const int t   = row & (TSEQ - 1);
    const int tok = idx[row];
    const float* te = tokw + (size_t)tok * DMODEL;
    const float* pe = posw + (size_t)t * DMODEL;

    float v[4];
    float s = 0.f, s2 = 0.f;
#pragma unroll
    for (int i = 0; i < 4; ++i) {
        int j = threadIdx.x + i * 256;
        float x = te[j] + pe[j];
        v[i] = x; s += x; s2 += x * x;
    }
    s_sum[threadIdx.x] = s; s_sq[threadIdx.x] = s2;
    __syncthreads();
    for (int st = 128; st > 0; st >>= 1) {
        if (threadIdx.x < st) {
            s_sum[threadIdx.x] += s_sum[threadIdx.x + st];
            s_sq[threadIdx.x]  += s_sq[threadIdx.x + st];
        }
        __syncthreads();
    }
    const float mean = s_sum[0] * (1.0f / DMODEL);
    const float var  = s_sq[0] * (1.0f / DMODEL) - mean * mean;
    const float rstd = rsqrtf(var + 1e-5f);

    unsigned short* orow = xb + (size_t)row * DMODEL;
#pragma unroll
    for (int i = 0; i < 4; ++i) {
        int j = threadIdx.x + i * 256;
        float xn = (v[i] - mean) * rstd * g[j] + bln[j];
        orow[j] = f2bf(xn);
    }
}

// ---------------------------------------------------------------------------
// 3/5) bf16 WMMA GEMM:  out[M,N] = A[M,K] * W[N,K]^T (+bias)
//      block = 256 thr (8 waves, 2x4), block tile 64(M) x 256(N),
//      wave tile 32(M) x 64(N): 8 accumulators, 2 A-frags x 4 B-frags
//      -> 8 WMMAs per 12 b128 loads per K-step of 32.
//      Requires M%64==0, N%256==0, K%32==0.
// ---------------------------------------------------------------------------
__global__ __launch_bounds__(256) void gemm_bf16_nt_kernel(const unsigned short* __restrict__ A,
                                                           const unsigned short* __restrict__ W,
                                                           const float* __restrict__ bias,
                                                           float* __restrict__ out,
                                                           int M, int N, int K) {
    const int lane  = threadIdx.x & 31;
    const int wave  = threadIdx.x >> 5;
    const int half  = lane >> 4;
    const int l16   = lane & 15;

    const int m0 = blockIdx.y * 64  + (wave >> 2) * 32;
    const int n0 = blockIdx.x * 256 + (wave & 3) * 64;

    const unsigned short* ap[2];
    ap[0] = A + (size_t)(m0 + l16) * K + half * 16;
    ap[1] = ap[0] + (size_t)16 * K;
    const unsigned short* bp[4];
#pragma unroll
    for (int j = 0; j < 4; ++j)
        bp[j] = W + (size_t)(n0 + j * 16 + l16) * K + half * 16;

    v8f acc[2][4];
#pragma unroll
    for (int i = 0; i < 2; ++i)
#pragma unroll
        for (int j = 0; j < 4; ++j) acc[i][j] = (v8f){};

    for (int k = 0; k < K; k += 32) {
#pragma unroll
        for (int j = 0; j < 4; ++j) __builtin_prefetch(bp[j] + k + 256, 0, 1);
        v16bf a[2], b[4];
#pragma unroll
        for (int i = 0; i < 2; ++i) a[i] = load_fragA(ap[i] + k);
#pragma unroll
        for (int j = 0; j < 4; ++j) b[j] = load_fragB(bp[j] + k);
#pragma unroll
        for (int i = 0; i < 2; ++i)
#pragma unroll
            for (int j = 0; j < 4; ++j)
                acc[i][j] = __builtin_amdgcn_wmma_f32_16x16x32_bf16(
                    false, a[i], false, b[j], (short)0, acc[i][j], false, false);
    }

#pragma unroll
    for (int j = 0; j < 4; ++j) {
        const int n = n0 + j * 16 + l16;
        const float bv = bias ? bias[n] : 0.f;
#pragma unroll
        for (int r = 0; r < 8; ++r) {
            const int m = m0 + r + half * 8;
            out[(size_t)m * N + n]        = acc[0][j][r] + bv;
            out[(size_t)(m + 16) * N + n] = acc[1][j][r] + bv;
        }
    }
}

// ---------------------------------------------------------------------------
// 4) leak-scan over T per (b,d) column, fused exact GELU, bf16 outputs
// ---------------------------------------------------------------------------
__global__ __launch_bounds__(256) void scan_gelu_kernel(const float* __restrict__ y,
                                                        unsigned short* __restrict__ Cb,
                                                        unsigned short* __restrict__ Gb) {
    const int tid = blockIdx.x * 256 + threadIdx.x;   // 0 .. BATCH*DMODEL-1
    const int b = tid >> 10;
    const int d = tid & (DMODEL - 1);
    size_t p = (size_t)b * TSEQ * DMODEL + d;
    float c = 0.f;
    for (int t = 0; t < TSEQ; ++t, p += DMODEL) {
        c = LEAK * c + y[p];
        Cb[p] = f2bf(c);
        float gl = 0.5f * c * (1.0f + erff(c * 0.70710678118654752f));
        Gb[p] = f2bf(gl);
    }
}

// ---------------------------------------------------------------------------
// launch
// ---------------------------------------------------------------------------
extern "C" void kernel_launch(void* const* d_in, const int* in_sizes, int n_in,
                              void* d_out, int out_size, void* d_ws, size_t ws_size,
                              hipStream_t stream) {
    (void)in_sizes; (void)n_in; (void)out_size; (void)ws_size;

    const int*   idx  = (const int*)d_in[0];
    const float* tokw = (const float*)d_in[1];
    const float* posw = (const float*)d_in[2];
    const float* ln_g = (const float*)d_in[3];
    const float* ln_b = (const float*)d_in[4];
    const float* Wc   = (const float*)d_in[5];
    const float* bc   = (const float*)d_in[6];
    const float* W0   = (const float*)d_in[7];
    const float* W1   = (const float*)d_in[8];
    float* out = (float*)d_out;

    // workspace layout (all chunks 256B-aligned)
    char* ws = (char*)d_ws;
    constexpr size_t NBIG = (size_t)VOCAB * DMODEL;        // 32,768,000
    constexpr size_t NACT = (size_t)MTOT * DMODEL;         //  2,097,152
    unsigned short* tokb = (unsigned short*)(ws);
    unsigned short* w0b  = (unsigned short*)(ws + NBIG * 2);
    unsigned short* w1b  = (unsigned short*)(ws + NBIG * 4);
    unsigned short* wcb  = (unsigned short*)(ws + NBIG * 6);
    unsigned short* xb   = (unsigned short*)(ws + NBIG * 6 + (size_t)DMODEL * DMODEL * 2);
    float*          ybuf = (float*)         (ws + NBIG * 6 + (size_t)DMODEL * DMODEL * 2 + NACT * 2);
    unsigned short* cb   = (unsigned short*)(ws + NBIG * 6 + (size_t)DMODEL * DMODEL * 2 + NACT * 2 + NACT * 4);
    unsigned short* gb   = (unsigned short*)(ws + NBIG * 6 + (size_t)DMODEL * DMODEL * 2 + NACT * 2 + NACT * 4 + NACT * 2);

    // 1) weight down-conversion to bf16
    cvt_bf16_kernel<<<2048, 256, 0, stream>>>(tokw, tokb, (int)NBIG);
    cvt_bf16_kernel<<<2048, 256, 0, stream>>>(W0,   w0b,  (int)NBIG);
    cvt_bf16_kernel<<<2048, 256, 0, stream>>>(W1,   w1b,  (int)NBIG);
    cvt_bf16_kernel<<<512,  256, 0, stream>>>(Wc,   wcb,  DMODEL * DMODEL);

    // 2) embedding + layernorm
    embed_ln_kernel<<<MTOT, 256, 0, stream>>>(idx, tokw, posw, ln_g, ln_b, xb);

    // 3) y = xhat @ Wc^T + bc
    gemm_bf16_nt_kernel<<<dim3(DMODEL / 256, MTOT / 64), 256, 0, stream>>>(
        xb, wcb, bc, ybuf, MTOT, DMODEL, DMODEL);

    // 4) linear recurrence + gelu, to bf16
    scan_gelu_kernel<<<(BATCH * DMODEL) / 256, 256, 0, stream>>>(ybuf, cb, gb);

    // 5) three readout GEMMs
    const size_t osz = (size_t)MTOT * VOCAB;
    gemm_bf16_nt_kernel<<<dim3(VOCAB / 256, MTOT / 64), 256, 0, stream>>>(
        gb, tokb, nullptr, out, MTOT, VOCAB, DMODEL);
    gemm_bf16_nt_kernel<<<dim3(VOCAB / 256, MTOT / 64), 256, 0, stream>>>(
        cb, w0b, nullptr, out + osz, MTOT, VOCAB, DMODEL);
    gemm_bf16_nt_kernel<<<dim3(VOCAB / 256, MTOT / 64), 256, 0, stream>>>(
        cb, w1b, nullptr, out + 2 * osz, MTOT, VOCAB, DMODEL);
}
